// MultiHeadSelfAttention_40080634807018
// MI455X (gfx1250) — matrix operand order repaired and checked by reading, not compiled, not run on hardware
//
#include <hip/hip_runtime.h>
#include <hip/hip_bf16.h>
#include <stdint.h>

typedef _Float16 h16v __attribute__((ext_vector_type(16)));
typedef _Float16 h8v  __attribute__((ext_vector_type(8)));
typedef _Float16 h4v  __attribute__((ext_vector_type(4)));
typedef float    f8v  __attribute__((ext_vector_type(8)));

#define DEV __device__ __forceinline__

DEV h8v ldh8(const _Float16* p) { return *(const h8v*)p; }
DEV h16v cat8(h8v a, h8v b) {
  return __builtin_shufflevector(a, b, 0,1,2,3,4,5,6,7,8,9,10,11,12,13,14,15);
}
DEV f8v fzero() { f8v z;
#pragma unroll
  for (int r = 0; r < 8; ++r) z[r] = 0.0f; return z; }
DEV uint32_t packh2(float a, float b) {
  union { _Float16 h[2]; uint32_t u; } t;
  t.h[0] = (_Float16)a; t.h[1] = (_Float16)b; return t.u;
}

// ---------------- f32 -> f16 conversion (n % 4 == 0) ----------------
__global__ void cvt_f16(const float* __restrict__ in, _Float16* __restrict__ out, int n) {
  int i = (blockIdx.x * blockDim.x + threadIdx.x) * 4;
  if (i < n) {
    float4 v = *(const float4*)(in + i);
    h4v o; o[0] = (_Float16)v.x; o[1] = (_Float16)v.y; o[2] = (_Float16)v.z; o[3] = (_Float16)v.w;
    *(h4v*)(out + i) = o;
  }
}

// ---------------- GEMM: C[M,N] = A[M,K] * W[N,K]^T (f16 in, f32 acc) ----------------
DEV void load_frags(const _Float16* __restrict__ A, const _Float16* __restrict__ W,
                    int m0, int n0, int K, int lidx, int lhalf, int k0,
                    h16v af[2], h16v bf[4]) {
#pragma unroll
  for (int mi = 0; mi < 2; ++mi) {
    // A 16x32 fragment: lane<16 rows m, k-chunks [0..7],[16..23]; lane>=16 [8..15],[24..31]
    const _Float16* p = A + (size_t)(m0 + mi*16 + lidx) * K + k0 + lhalf*8;
    af[mi] = cat8(ldh8(p), ldh8(p + 16));
  }
#pragma unroll
  for (int ni = 0; ni < 4; ++ni) {
    // B 32x16 fragment: lane<16 col n, k-chunks [0..7],[16..23]; lane>=16 [8..15],[24..31]
    const _Float16* p = W + (size_t)(n0 + ni*16 + lidx) * K + k0 + lhalf*8;
    bf[ni] = cat8(ldh8(p), ldh8(p + 16));
  }
}

DEV void mma8(const h16v af[2], const h16v bf[4], f8v acc[2][4]) {
#pragma unroll
  for (int mi = 0; mi < 2; ++mi)
#pragma unroll
    for (int ni = 0; ni < 4; ++ni)
      acc[mi][ni] = __builtin_amdgcn_wmma_f32_16x16x32_f16(
          false, af[mi], false, bf[ni], (short)0, acc[mi][ni], false, false);
}

// MODE 0: store f16 row-major [M,N]
// MODE 1: store f16 transposed-per-head: Vt[b,h,d,s], row pitch SP
// MODE 2: store f32 row-major [M,N]
template <int MODE>
__global__ __launch_bounds__(256) void gemm16(
    const _Float16* __restrict__ A, const _Float16* __restrict__ W,
    void* __restrict__ out, int M, int N, int K, int S, int SP)
{
  const int lane  = threadIdx.x & 31;
  const int wave  = threadIdx.x >> 5;
  const int lidx  = lane & 15;
  const int lhalf = lane >> 4;
  const int m0 = blockIdx.x * 64  + (wave & 1) * 32;
  const int n0 = blockIdx.y * 256 + (wave >> 1) * 64;

  f8v acc[2][4];
#pragma unroll
  for (int mi = 0; mi < 2; ++mi)
#pragma unroll
    for (int ni = 0; ni < 4; ++ni) acc[mi][ni] = fzero();

  // software-pipelined k loop (double-buffered fragments)
  h16v af0[2], bf0[4], af1[2], bf1[4];
  load_frags(A, W, m0, n0, K, lidx, lhalf, 0, af0, bf0);
  int k0 = 0;
  for (; k0 + 64 < K; k0 += 64) {
    load_frags(A, W, m0, n0, K, lidx, lhalf, k0 + 32, af1, bf1);
    mma8(af0, bf0, acc);
    load_frags(A, W, m0, n0, K, lidx, lhalf, k0 + 64, af0, bf0);
    mma8(af1, bf1, acc);
  }
  load_frags(A, W, m0, n0, K, lidx, lhalf, k0 + 32, af1, bf1);
  mma8(af0, bf0, acc);
  mma8(af1, bf1, acc);

#pragma unroll
  for (int mi = 0; mi < 2; ++mi) {
#pragma unroll
    for (int ni = 0; ni < 4; ++ni) {
      const int n  = n0 + ni*16 + lidx;
      const int mb = m0 + mi*16 + lhalf*8;
      if (MODE == 0) {
        _Float16* C = (_Float16*)out;
#pragma unroll
        for (int r = 0; r < 8; ++r) C[(size_t)(mb + r) * N + n] = (_Float16)acc[mi][ni][r];
      } else if (MODE == 2) {
        float* C = (float*)out;
#pragma unroll
        for (int r = 0; r < 8; ++r) C[(size_t)(mb + r) * N + n] = acc[mi][ni][r];
      } else { // MODE 1: Vt[b,h,d,s]
        _Float16* Vt = (_Float16*)out;
        const int b = mb / S, s = mb - b * S;
        const int h = n >> 6, d = n & 63;
        h8v v;
#pragma unroll
        for (int r = 0; r < 8; ++r) v[r] = (_Float16)acc[mi][ni][r];
        *(h8v*)(Vt + (((size_t)(b * 16 + h) * 64 + d) * SP + s)) = v;
      }
    }
  }
}

// ---------------- Flash attention, one 16-query tile per wave ----------------
DEV void loadK(const _Float16* __restrict__ kbase, int kk, int Dm, int lidx, int lhalf,
               h16v klo[2], h16v khi[2]) {
#pragma unroll
  for (int c = 0; c < 2; ++c) {
    const _Float16* kp = kbase + (size_t)(kk + c*16 + lidx) * Dm;
    klo[c] = cat8(ldh8(kp +      lhalf*8), ldh8(kp + 16 + lhalf*8));
    khi[c] = cat8(ldh8(kp + 32 + lhalf*8), ldh8(kp + 48 + lhalf*8));
  }
}

__global__ __launch_bounds__(256) void attn_fwd(
    const _Float16* __restrict__ Qh, const _Float16* __restrict__ Kh,
    const _Float16* __restrict__ Vt, _Float16* __restrict__ Oh,
    int S, int SP)
{
  const int Dm = 1024;
  const int lane  = threadIdx.x & 31;
  const int wave  = threadIdx.x >> 5;
  const int lidx  = lane & 15;
  const int lhalf = lane >> 4;

  const int tilesPerBH = S >> 4;                  // 128
  const int qtg = blockIdx.x * 8 + wave;
  const int bh  = qtg / tilesPerBH;
  const int qt  = qtg - bh * tilesPerBH;
  const int b   = bh >> 4;
  const int h   = bh & 15;
  const int qbase = qt << 4;

  // Q as B-operand fragments (Q^T dk-slices 0..31 and 32..63), pre-scaled by 1/sqrt(dk)
  const _Float16* qrow = Qh + (size_t)(b * S + qbase + lidx) * Dm + h * 64;
  h16v qf[2];
#pragma unroll
  for (int i = 0; i < 2; ++i) {
    const _Float16* p = qrow + i*32 + lhalf*8;
    qf[i] = cat8(ldh8(p), ldh8(p + 16));
#pragma unroll
    for (int e = 0; e < 16; ++e) qf[i][e] = qf[i][e] * (_Float16)0.125f; // exact (power of 2)
  }

  const _Float16* kbase = Kh + (size_t)(b * S) * Dm + h * 64;
  const _Float16* vbase = Vt + (size_t)bh * 64 * SP;

  float mrow = -__builtin_huge_valf();
  float rsum = 0.0f;
  f8v oacc[4];
#pragma unroll
  for (int j = 0; j < 4; ++j) oacc[j] = fzero();

  const int qa   = qbase + lidx;   // absolute query for this lane's column
  const int qend = qbase + 16;

  // pipelined K fragments
  h16v klo[2], khi[2], nklo[2], nkhi[2];
  loadK(kbase, 0, Dm, lidx, lhalf, klo, khi);

  for (int kk = 0; kk < qend; kk += 32) {
    // V fragments for this step (independent of score chain)
    h16v vf[4];
#pragma unroll
    for (int j = 0; j < 4; ++j) {
      const _Float16* vp = vbase + (size_t)(j*16 + lidx) * SP + kk + lhalf*16;
      vf[j] = cat8(ldh8(vp), ldh8(vp + 8));
    }

    // S^T tiles (16 keys x 16 queries), c = key sub-tile
    f8v sc[2];
#pragma unroll
    for (int c = 0; c < 2; ++c) {
      f8v ca = fzero();
      ca = __builtin_amdgcn_wmma_f32_16x16x32_f16(false, klo[c], false, qf[0], (short)0, ca, false, false);
      ca = __builtin_amdgcn_wmma_f32_16x16x32_f16(false, khi[c], false, qf[1], (short)0, ca, false, false);
      sc[c] = ca;
    }

    // prefetch next step's K fragments (unconditional; overrun stays inside ws, values unused)
    loadK(kbase, kk + 32, Dm, lidx, lhalf, nklo, nkhi);

    // online softmax (mask only needed on the final key-step)
    float pv[16];
    float tmax = -__builtin_huge_valf();
    if (kk + 48 <= qend) {          // fully unmasked step
#pragma unroll
      for (int c = 0; c < 2; ++c)
#pragma unroll
        for (int r = 0; r < 8; ++r) {
          const float v = sc[c][r];
          pv[c*8 + r] = v;
          tmax = fmaxf(tmax, v);
        }
    } else {                        // causal-masked step
#pragma unroll
      for (int c = 0; c < 2; ++c)
#pragma unroll
        for (int r = 0; r < 8; ++r) {
          const int key = kk + c*16 + lhalf*8 + r;
          const float v = (key <= qa) ? sc[c][r] : -__builtin_huge_valf();
          pv[c*8 + r] = v;
          tmax = fmaxf(tmax, v);
        }
    }
    tmax = fmaxf(tmax, __shfl_xor(tmax, 16, 32));
    const float mnew  = fmaxf(mrow, tmax);
    const float alpha = __expf(mrow - mnew);
    float psum = 0.0f;
#pragma unroll
    for (int i = 0; i < 16; ++i) { pv[i] = __expf(pv[i] - mnew); psum += pv[i]; }
    psum += __shfl_xor(psum, 16, 32);
    rsum = rsum * alpha + psum;
    mrow = mnew;

    // build P^T B-fragment (32 keys x 16 queries) via lane^16 swaps
    uint32_t pk0[4], pk1[4], o0[4], o1[4];
#pragma unroll
    for (int j2 = 0; j2 < 4; ++j2) {
      pk0[j2] = packh2(pv[2*j2],     pv[2*j2 + 1]);
      pk1[j2] = packh2(pv[8 + 2*j2], pv[8 + 2*j2 + 1]);
      o0[j2]  = (uint32_t)__shfl_xor((int)pk0[j2], 16, 32);
      o1[j2]  = (uint32_t)__shfl_xor((int)pk1[j2], 16, 32);
    }
    union { h16v h; uint32_t u[8]; } bp;
#pragma unroll
    for (int j2 = 0; j2 < 4; ++j2) {
      bp.u[j2]     = lhalf ? o1[j2]  : pk0[j2];
      bp.u[j2 + 4] = lhalf ? pk1[j2] : o0[j2];
    }

    // O^T += V^T-slice x P^T, with rescale by alpha
#pragma unroll
    for (int j = 0; j < 4; ++j) {
      f8v t = oacc[j];
#pragma unroll
      for (int r = 0; r < 8; ++r) t[r] *= alpha;
      oacc[j] = __builtin_amdgcn_wmma_f32_16x16x32_f16(false, vf[j], false, bp.h, (short)0, t, false, false);
    }

    // rotate pipelined K fragments
#pragma unroll
    for (int c = 0; c < 2; ++c) { klo[c] = nklo[c]; khi[c] = nkhi[c]; }
  }

  const float inv = 1.0f / rsum;
  _Float16* orow = Oh + (size_t)(b * S + qbase + lidx) * Dm + h * 64;
#pragma unroll
  for (int j = 0; j < 4; ++j) {
    h8v o;
#pragma unroll
    for (int r = 0; r < 8; ++r) o[r] = (_Float16)(oacc[j][r] * inv);
    *(h8v*)(orow + j*16 + lhalf*8) = o;
  }
}

// ---------------- launch ----------------
extern "C" void kernel_launch(void* const* d_in, const int* in_sizes, int n_in,
                              void* d_out, int out_size, void* d_ws, size_t ws_size,
                              hipStream_t stream) {
  (void)in_sizes; (void)n_in; (void)out_size; (void)ws_size;
  const int B = 4, S = 2048, Dm = 1024;
  const int M = B * S;           // 8192
  const int SP = S + 64;         // padded Vt pitch (covers causal-edge reads)

  const float* x  = (const float*)d_in[0];
  const float* Wq = (const float*)d_in[1];
  const float* Wk = (const float*)d_in[2];
  const float* Wv = (const float*)d_in[3];
  const float* Wo = (const float*)d_in[4];

  char* ws = (char*)d_ws;
  const size_t MB = 1u << 20;
  _Float16* xh  = (_Float16*)(ws + 0 * MB);   // 16 MiB
  _Float16* wqh = (_Float16*)(ws + 16 * MB);  // 2 MiB
  _Float16* wkh = (_Float16*)(ws + 18 * MB);
  _Float16* wvh = (_Float16*)(ws + 20 * MB);
  _Float16* woh = (_Float16*)(ws + 22 * MB);
  _Float16* qh  = (_Float16*)(ws + 24 * MB);  // 16 MiB
  _Float16* kh  = (_Float16*)(ws + 40 * MB);  // 16 MiB
  _Float16* vt  = (_Float16*)(ws + 56 * MB);  // B*16*64*SP*2 ~= 16.5 MiB
  _Float16* oh  = (_Float16*)(ws + 92 * MB);  // 16 MiB

  const int nx = M * Dm, nw = Dm * Dm;
  cvt_f16<<<(nx/4 + 255)/256, 256, 0, stream>>>(x,  xh,  nx);
  cvt_f16<<<(nw/4 + 255)/256, 256, 0, stream>>>(Wq, wqh, nw);
  cvt_f16<<<(nw/4 + 255)/256, 256, 0, stream>>>(Wk, wkh, nw);
  cvt_f16<<<(nw/4 + 255)/256, 256, 0, stream>>>(Wv, wvh, nw);
  cvt_f16<<<(nw/4 + 255)/256, 256, 0, stream>>>(Wo, woh, nw);

  dim3 gg(M / 64, Dm / 256);
  gemm16<0><<<gg, 256, 0, stream>>>(xh, wqh, qh, M, Dm, Dm, S, SP);
  gemm16<0><<<gg, 256, 0, stream>>>(xh, wkh, kh, M, Dm, Dm, S, SP);
  gemm16<1><<<gg, 256, 0, stream>>>(xh, wvh, vt, M, Dm, Dm, S, SP);

  const int nTiles = B * 16 * (S / 16);       // 8192 wave-tiles
  attn_fwd<<<nTiles / 8, 256, 0, stream>>>(qh, kh, vt, oh, S, SP);

  gemm16<2><<<gg, 256, 0, stream>>>(oh, woh, d_out, M, Dm, Dm, S, SP);
}
